// LoRAMlp_36429912605326
// MI455X (gfx1250) — matrix operand order repaired and checked by reading, not compiled, not run on hardware
//
#include <hip/hip_runtime.h>
#include <hip/hip_bf16.h>
#include <math.h>

// ---------------------------------------------------------------------------
// LoRA-MLP for MI455X (gfx1250, wave32, WMMA + async global->LDS copies).
//   M = B*N = 64*197 = 12608 tokens, D = 768, H = 3072, R = 16, K = 1.
// Strategy:
//   1) Fold LoRA into dense weights in fp32:  W1e = W1 + B1*A1,  W2e = W2 + B2*A2.
//   2) Cast x / W1e / W2e to bf16 (fp32 accumulate in WMMA).
//   3) GEMM1 (+bias, exact-erf GELU) -> h bf16;  GEMM2 (+bias) -> out fp32.
// Data movement: k-chunks staged into LDS with GLOBAL_LOAD_ASYNC_TO_LDS_B128
// (ASYNCcnt), double-buffered with the k-loop unrolled x2 so buffer indices
// are compile-time constants; fragments read back with ds_load_b128 in the
// ISA-specified WMMA register layouts.
// Roofline: ~119 GFLOP vs ~0.22 GB HBM -> near memory roofline with bf16 WMMA.
// ---------------------------------------------------------------------------

#define Mtok 12608
#define Ddim 768
#define Hdim 3072
#define Rrk  16

typedef __attribute__((ext_vector_type(16))) __bf16 bf16x16;
typedef __attribute__((ext_vector_type(8)))  __bf16 bf16x8;
typedef __attribute__((ext_vector_type(8)))  float  f32x8;

// Async 16B copy global -> LDS, tracked by ASYNCcnt (CDNA5 ISA).
// lds_off: wave-relative LDS byte address (low 32 bits of generic pointer).
__device__ __forceinline__ void async_copy_b128(unsigned lds_off, const void* gptr) {
    asm volatile("global_load_async_to_lds_b128 %0, %1, off"
                 :: "v"(lds_off), "v"((unsigned long long)(uintptr_t)gptr)
                 : "memory");
}
__device__ __forceinline__ void async_wait_all() {
    asm volatile("s_wait_asynccnt 0x0" ::: "memory");
}

// ---------------------------------------------------------------------------
// Main GEMM: C[M, NDIM] = A[M, KDIM] * Bw[NDIM, KDIM]^T  (+bias, +epilogue)
// Block = 256 threads = 8 waves (2 along M x 4 along N); wave tile 32x64
// (2x4 WMMA tiles); block tile 64x256; k-chunk 32, LDS double-buffered.
// LDS rows padded to 40 bf16 (80B = 20 banks) -> conflict-free b128 reads.
// ---------------------------------------------------------------------------
template<int KDIM, int NDIM, bool GELU>
__global__ __launch_bounds__(256)
void gemm_bias_kernel(const __bf16* __restrict__ A,
                      const __bf16* __restrict__ Bw,
                      const float*  __restrict__ bias,
                      __bf16*       __restrict__ outBf,
                      float*        __restrict__ outF32) {
    static_assert((KDIM / 32) % 2 == 0, "k-chunk count must be even");

    __shared__ __bf16 As[2][64][40];     //  10 KB
    __shared__ __bf16 Bs[2][256][40];    //  40 KB

    const int t    = threadIdx.x;
    const int lane = t & 31;
    const int wid  = t >> 5;
    const int wm   = wid & 1;        // 2 waves along M
    const int wn   = wid >> 1;       // 4 waves along N
    const int hi   = lane >> 4;      // lane half
    const int lid  = lane & 15;

    // ---- staging: thread t owns row rS (0..63), 16B chunk cS of each tile ----
    const int rS = t >> 2;           // 0..63
    const int cS = (t & 3) * 8;      // element col: 0,8,16,24

    // persistent global pointers, each advanced 32 elements (64B) per k-chunk
    const __bf16* gA  = A  + (size_t)(blockIdx.x * 64  + rS) * KDIM + cS;
    const __bf16* gB0 = Bw + (size_t)(blockIdx.y * 256 + rS) * KDIM + cS;
    const __bf16* gB1 = gB0 + (size_t)64  * KDIM;
    const __bf16* gB2 = gB0 + (size_t)128 * KDIM;
    const __bf16* gB3 = gB0 + (size_t)192 * KDIM;

    unsigned aDst[2], bDst[2][4];
#pragma unroll
    for (int p = 0; p < 2; ++p) {
        aDst[p] = (unsigned)(uintptr_t)&As[p][rS][cS];
#pragma unroll
        for (int q = 0; q < 4; ++q)
            bDst[p][q] = (unsigned)(uintptr_t)&Bs[p][rS + q * 64][cS];
    }

    // stage one 32-wide k-chunk into LDS buffer p, advance global pointers
    auto stage = [&](int p) {
        async_copy_b128(aDst[p], gA);
        async_copy_b128(bDst[p][0], gB0);
        async_copy_b128(bDst[p][1], gB1);
        async_copy_b128(bDst[p][2], gB2);
        async_copy_b128(bDst[p][3], gB3);
        gA += 32; gB0 += 32; gB1 += 32; gB2 += 32; gB3 += 32;
    };

    f32x8 acc[2][4];
    const f32x8 zero = {0.f, 0.f, 0.f, 0.f, 0.f, 0.f, 0.f, 0.f};
#pragma unroll
    for (int i = 0; i < 2; ++i)
#pragma unroll
        for (int j = 0; j < 4; ++j)
            acc[i][j] = zero;

    // fragment readback + 8 WMMAs from LDS buffer p
    auto compute = [&](int p) {
        // A-fragment (16x32, MxK): lane (m = lid) holds K = kb..kb+7, kb+16..kb+23,
        // kb = hi*8  -> two aligned ds_load_b128.
        bf16x16 af[2];
#pragma unroll
        for (int i = 0; i < 2; ++i) {
            const __bf16* pa = &As[p][wm * 32 + i * 16 + lid][hi * 8];
            union { bf16x16 f; bf16x8 h[2]; } u;
            u.h[0] = *(const bf16x8*)(pa);
            u.h[1] = *(const bf16x8*)(pa + 16);
            af[i] = u.f;
        }
        // B-fragment (32x16, KxN): lane (n = lid) holds K = kb..kb+7, kb+16..kb+23,
        // kb = hi*8 -> two aligned ds_load_b128.
        bf16x16 bfr[4];
#pragma unroll
        for (int j = 0; j < 4; ++j) {
            const __bf16* pb = &Bs[p][wn * 64 + j * 16 + lid][hi * 8];
            union { bf16x16 f; bf16x8 h[2]; } u;
            u.h[0] = *(const bf16x8*)(pb);
            u.h[1] = *(const bf16x8*)(pb + 16);
            bfr[j] = u.f;
        }
#pragma unroll
        for (int i = 0; i < 2; ++i)
#pragma unroll
            for (int j = 0; j < 4; ++j)
                acc[i][j] = __builtin_amdgcn_wmma_f32_16x16x32_bf16(
                    /*neg_a=*/false, af[i], /*neg_b=*/false, bfr[j],
                    /*c_mod=*/(short)0, acc[i][j],
                    /*reuse_a=*/false, /*reuse_b=*/false);
    };

    // ---- software pipeline, unrolled x2 so buffer index is a constant ----
    stage(0);
    async_wait_all();
    __syncthreads();

    for (int k0 = 0; k0 < KDIM; k0 += 64) {
        stage(1);                       // chunk k0+32 (always exists: even count)
        compute(0);                     // chunk k0
        async_wait_all();
        __syncthreads();

        if (k0 + 64 < KDIM)
            stage(0);                   // chunk k0+64
        compute(1);                     // chunk k0+32
        async_wait_all();
        __syncthreads();
    }

    const int m0 = blockIdx.x * 64  + wm * 32;
    const int n0 = blockIdx.y * 256 + wn * 64;

    // C layout: VGPR v, lanes 0-15 -> M = v, lanes 16-31 -> M = v + 8; N = lid.
#pragma unroll
    for (int j = 0; j < 4; ++j) {
        const int   nn = n0 + j * 16 + lid;
        const float bv = bias[nn];
#pragma unroll
        for (int i = 0; i < 2; ++i) {
            const int mbase = m0 + i * 16 + hi * 8;
#pragma unroll
            for (int v = 0; v < 8; ++v) {
                float val = acc[i][j][v] + bv;
                if (GELU) {
                    val = 0.5f * val * (1.0f + erff(val * 0.70710678118654752f));
                    outBf[(size_t)(mbase + v) * NDIM + nn] = (__bf16)val;
                } else {
                    outF32[(size_t)(mbase + v) * NDIM + nn] = val;
                }
            }
        }
    }
}

// ---------------------------------------------------------------------------
// Prep kernels
// ---------------------------------------------------------------------------
__global__ __launch_bounds__(256)
void cvt_bf16_kernel(const float* __restrict__ in, __bf16* __restrict__ out, int n4) {
    int i = (blockIdx.x * 256 + threadIdx.x) * 4;
    if (i < n4) {
        float4 v = *(const float4*)(in + i);
        out[i + 0] = (__bf16)v.x;
        out[i + 1] = (__bf16)v.y;
        out[i + 2] = (__bf16)v.z;
        out[i + 3] = (__bf16)v.w;
    }
}

// W1e[h,d] = W1[h,d] + sum_r B1[h,r] * A1[r,d]   (H*D threads)
__global__ __launch_bounds__(256)
void fold1_kernel(const float* __restrict__ W1, const float* __restrict__ A1,
                  const float* __restrict__ B1, __bf16* __restrict__ out) {
    const int idx = blockIdx.x * 256 + threadIdx.x;   // < H*D
    const int h = idx / Ddim;
    const int d = idx - h * Ddim;
    float acc = W1[idx];
#pragma unroll
    for (int r = 0; r < Rrk; ++r)
        acc += B1[h * Rrk + r] * A1[r * Ddim + d];
    out[idx] = (__bf16)acc;
}

// W2e[d,h] = W2[d,h] + sum_r B2[d,r] * A2[r,h]   (D*H threads)
__global__ __launch_bounds__(256)
void fold2_kernel(const float* __restrict__ W2, const float* __restrict__ A2,
                  const float* __restrict__ B2, __bf16* __restrict__ out) {
    const int idx = blockIdx.x * 256 + threadIdx.x;   // < D*H
    const int d = idx / Hdim;
    const int h = idx - d * Hdim;
    float acc = W2[idx];
#pragma unroll
    for (int r = 0; r < Rrk; ++r)
        acc += B2[d * Rrk + r] * A2[r * Hdim + h];
    out[idx] = (__bf16)acc;
}

// ---------------------------------------------------------------------------
// Launch
// ---------------------------------------------------------------------------
extern "C" void kernel_launch(void* const* d_in, const int* in_sizes, int n_in,
                              void* d_out, int out_size, void* d_ws, size_t ws_size,
                              hipStream_t stream) {
    const float* x  = (const float*)d_in[0];
    const float* W1 = (const float*)d_in[1];
    const float* b1 = (const float*)d_in[2];
    const float* A1 = (const float*)d_in[3];
    const float* B1 = (const float*)d_in[4];
    const float* W2 = (const float*)d_in[5];
    const float* b2 = (const float*)d_in[6];
    const float* A2 = (const float*)d_in[7];
    const float* B2 = (const float*)d_in[8];
    float* out = (float*)d_out;

    // Workspace layout (all 256B-aligned by construction):
    //   Xbf  [M*D]  bf16 : 19,365,888 B
    //   W1e  [H*D]  bf16 :  4,718,592 B
    //   W2e  [D*H]  bf16 :  4,718,592 B
    //   Hbf  [M*H]  bf16 : 77,463,552 B     total ~101 MB
    char* ws = (char*)d_ws;
    __bf16* Xbf = (__bf16*)(ws);
    __bf16* W1e = (__bf16*)(ws + (size_t)Mtok * Ddim * 2);
    __bf16* W2e = (__bf16*)(ws + (size_t)Mtok * Ddim * 2 + (size_t)Hdim * Ddim * 2);
    __bf16* Hbf = (__bf16*)(ws + (size_t)Mtok * Ddim * 2 + (size_t)Hdim * Ddim * 4);

    // 1) x -> bf16          (M*D = 9,682,944 elements, 4 per thread)
    cvt_bf16_kernel<<<(Mtok * Ddim) / (256 * 4), 256, 0, stream>>>(x, Xbf, Mtok * Ddim);

    // 2) fold LoRA into weights (fp32 math, bf16 store)
    fold1_kernel<<<(Hdim * Ddim) / 256, 256, 0, stream>>>(W1, A1, B1, W1e);
    fold2_kernel<<<(Ddim * Hdim) / 256, 256, 0, stream>>>(W2, A2, B2, W2e);

    // 3) fc1: [12608 x 768] * [3072 x 768]^T + b1, GELU -> Hbf
    dim3 g1(Mtok / 64, Hdim / 256);
    gemm_bias_kernel<Ddim, Hdim, true><<<g1, 256, 0, stream>>>(Xbf, W1e, b1, Hbf, nullptr);

    // 4) fc2: [12608 x 3072] * [768 x 3072]^T + b2 -> out (fp32)
    dim3 g2(Mtok / 64, Ddim / 256);
    gemm_bias_kernel<Hdim, Ddim, false><<<g2, 256, 0, stream>>>(Hbf, W2e, b2, nullptr, out);
}
